// GNN_QM9_pna_9328668966930
// MI455X (gfx1250) — compile-verified, matrix operand order checked by reading, not run on hardware
//
#include <hip/hip_runtime.h>
#include <hip/hip_bf16.h>
#include <math.h>

// ---------------------------------------------------------------------------
// Problem constants (from reference)
// ---------------------------------------------------------------------------
#define NN   10000
#define EE   160000
#define DD   128
#define EDD  64
#define TT   4
#define LL   3
#define FIN  128
#define FOUT 32
#define CH   40000   // edge chunk: 3 x CH x 256 x 4B = 123MB -> fits 192MB L2

typedef float v8f  __attribute__((ext_vector_type(8)));
typedef __bf16 v16bf __attribute__((ext_vector_type(16)));

// ---------------------------------------------------------------------------
// Generic multi-source WMMA GEMM:
//   C = act( [A0|A1|A2] @ W + bias )
// Each source s has its own leading dim and optional row-gather index array.
// Block tile: 128(M) x BN(N); 256 threads = 8 wave32 waves arranged 4x2.
//   BN=64: each wave owns a 32x32 C sub-tile -> 4 v_wmma per 32-wide K step.
//   BN=32: each wave owns 32x16 -> 2 v_wmma (only for Nc==32 GEMMs).
// A/B staged through LDS as packed bf16 pairs matching the ISA 16-bit
// A-matrix VGPR layout; accumulate f32.
// Steady-state K loop is branch-free: gather indices / row pointers / W
// pointers are all K-invariant and hoisted; out-of-range M rows clamp to
// row 0 (their C stores are masked), so no guards remain in the hot loop.
// Requires: all K_s multiples of 32, Nc multiple of BN (true here).
// ---------------------------------------------------------------------------
struct GemmParams {
  const float* A0; const int* g0; int ld0; int K0;
  const float* A1; const int* g1; int ld1; int K1;
  const float* A2; const int* g2; int ld2; int K2;
  const float* W;  int ldw;
  const float* bias;
  float* C; int ldc;
  int M; int relu;
};

__device__ __forceinline__ unsigned int pk_bf16(float a, float b) {
  // low 16 = bf16(a) (even k), high 16 = bf16(b) (odd k) per ISA VGPR layout
  return (__float_as_uint(a) >> 16) | (__float_as_uint(b) & 0xffff0000u);
}

template <int BN>
__global__ __launch_bounds__(256) void wmma_gemm(GemmParams p) {
  constexpr int NFRAG = BN / 32;       // B fragments (16 cols each) per wave
  __shared__ unsigned int lA[128][17]; // [row][kpair] bf16x2, padded
  __shared__ unsigned int lB[BN][17];  // [col][kpair] bf16x2 (B transposed)

  const int tid  = threadIdx.x;
  const int lane = tid & 31;
  const int wv   = tid >> 5;     // 0..7
  const int wm   = wv >> 1;      // 0..3 : M sub-block of 32 rows
  const int wn   = wv & 1;       // 0..1 : N sub-block of BN/2 cols
  const int hl   = lane >> 4;    // half-wave selects K sub-range
  const int mr   = lane & 15;    // row (A) / col (B) within 16
  const int rowBase = blockIdx.y * 128;
  const int colBase = blockIdx.x * BN;

  // K-invariant W base pointers for the B-tile stage (per thread)
  const float* wbase[BN / 16];
#pragma unroll
  for (int q2 = 0; q2 < BN / 16; ++q2) {
    int q   = tid + q2 * 256;       // 0 .. BN*16-1
    int col = q >> 4;               // 0..BN-1
    int kp  = q & 15;               // 0..15
    wbase[q2] = p.W + (long)(kp * 2) * p.ldw + (colBase + col);
  }

  v8f acc0[NFRAG]; v8f acc1[NFRAG];
#pragma unroll
  for (int f = 0; f < NFRAG; ++f) { acc0[f] = {}; acc1[f] = {}; }

  int kbase = 0;
  for (int si = 0; si < 3; ++si) {
    const float* A; const int* g; int ld, Ks;
    if (si == 0)      { A = p.A0; g = p.g0; ld = p.ld0; Ks = p.K0; }
    else if (si == 1) { A = p.A1; g = p.g1; ld = p.ld1; Ks = p.K1; }
    else              { A = p.A2; g = p.g2; ld = p.ld2; Ks = p.K2; }
    if (Ks == 0) continue;

    // ---- hoist per-thread A row pointers (K-invariant): gather index is
    // loaded ONCE per source; out-of-range rows clamp to row 0 (masked at
    // the C store), keeping the steady-state loop branch-free.
    const float* aptr[4];
#pragma unroll
    for (int i = 0; i < 4; ++i) {
      int lin = tid * 4 + i;          // 0..1023
      int row = lin >> 3;             // 0..127
      int kq  = (lin & 7) << 2;       // 0,4,...,28
      int gr  = rowBase + row;
      int grc = (gr < p.M) ? gr : 0;
      long ri = g ? (long)g[grc] : (long)grc;
      aptr[i] = A + ri * (long)ld + kq;
    }

    for (int kl = 0; kl < Ks; kl += 32) {   // branch-free steady state
      const int kb = kbase + kl;            // global K row into W

      __syncthreads();
      // ---- stage A tile: 128 rows x 32 k, float4 loads -> packed bf16
#pragma unroll
      for (int i = 0; i < 4; ++i) {
        int lin = tid * 4 + i;
        int row = lin >> 3;
        int kq  = (lin & 7) << 2;
        float4 v = *(const float4*)(aptr[i] + kl);
        lA[row][(kq >> 1) + 0] = pk_bf16(v.x, v.y);
        lA[row][(kq >> 1) + 1] = pk_bf16(v.z, v.w);
      }
      // ---- stage B tile: 32 k x BN cols of W, transposed to [col][kpair]
#pragma unroll
      for (int q2 = 0; q2 < BN / 16; ++q2) {
        int q   = tid + q2 * 256;
        int col = q >> 4;
        int kp  = q & 15;
        const float* wp = wbase[q2] + (long)kb * p.ldw;
        lB[col][kp] = pk_bf16(wp[0], wp[p.ldw]);
      }
      __syncthreads();

      // ---- fragments per ISA 16-bit A-matrix layout
      union { v16bf v; unsigned int u[8]; } a0, a1, b[NFRAG];
#pragma unroll
      for (int q = 0; q < 8; ++q) {
        int kp = (q < 4) ? (hl * 4 + q) : (8 + hl * 4 + (q - 4));
        a0.u[q] = lA[wm * 32 + mr][kp];
        a1.u[q] = lA[wm * 32 + 16 + mr][kp];
#pragma unroll
        for (int f = 0; f < NFRAG; ++f)
          b[f].u[q] = lB[wn * (BN / 2) + f * 16 + mr][kp];
      }
#pragma unroll
      for (int f = 0; f < NFRAG; ++f) {
        acc0[f] = __builtin_amdgcn_wmma_f32_16x16x32_bf16(
            false, a0.v, false, b[f].v, (short)0, acc0[f], false, false);
        acc1[f] = __builtin_amdgcn_wmma_f32_16x16x32_bf16(
            false, a1.v, false, b[f].v, (short)0, acc1[f], false, false);
      }
    }
    kbase += Ks;
  }

  // ---- epilogue: bias + optional ReLU; C/D layout VGPR r -> M = r + hl*8
#pragma unroll
  for (int f = 0; f < NFRAG; ++f) {
    const int gn = colBase + wn * (BN / 2) + f * 16 + mr;
    const float bv = p.bias ? p.bias[gn] : 0.f;
#pragma unroll
    for (int r = 0; r < 8; ++r) {
      int gm0 = rowBase + wm * 32 + hl * 8 + r;
      if (gm0 < p.M) {
        float v = acc0[f][r] + bv;
        if (p.relu) v = fmaxf(v, 0.f);
        p.C[(long)gm0 * p.ldc + gn] = v;
      }
      int gm1 = gm0 + 16;
      if (gm1 < p.M) {
        float v = acc1[f][r] + bv;
        if (p.relu) v = fmaxf(v, 0.f);
        p.C[(long)gm1 * p.ldc + gn] = v;
      }
    }
  }
}

// ---------------------------------------------------------------------------
// Elementwise / aggregation helper kernels
// ---------------------------------------------------------------------------
__global__ void fill_kernel(float* p, float v, long n) {
  long i = (long)blockIdx.x * blockDim.x + threadIdx.x;
  if (i < n) p[i] = v;
}

__global__ void count_kernel(const int* dst, float* cnt, int e) {
  int i = blockIdx.x * blockDim.x + threadIdx.x;
  if (i < e) atomicAdd(&cnt[dst[i]], 1.f);
}

__device__ __forceinline__ void atomicMaxF(float* a, float v) {
  unsigned int* ua = (unsigned int*)a;
  unsigned int cur = *ua;
  while (__uint_as_float(cur) < v) {
    unsigned int prev = atomicCAS(ua, cur, __float_as_uint(v));
    if (prev == cur) break;
    cur = prev;
  }
}

// fused PNA segment reduce: sum / sum-of-squares / max, per task t
__global__ void aggregate_kernel(const float* msg, const int* dst, int mc, int t,
                                 float* sum, float* sumsq, float* mx) {
  long i = (long)blockIdx.x * blockDim.x + threadIdx.x;
  long tot = (long)mc * FIN;
  if (i >= tot) return;
  int e = (int)(i >> 7), j = (int)(i & 127);
  int n = dst[e];
  float v = msg[i];
  long o = (((long)n * TT + t) << 7) + j;
  atomicAdd(&sum[o], v);
  atomicAdd(&sumsq[o], v * v);
  atomicMaxF(&mx[o], v);
}

// build per-node post-MLP input: [x | mean | max | std]
__global__ void build_postin_kernel(const float* x, const float* sum,
                                    const float* sumsq, const float* mx,
                                    const float* cnt, float* pin, int t) {
  long i = (long)blockIdx.x * blockDim.x + threadIdx.x;
  if (i >= (long)NN * FIN) return;
  int nd = (int)(i >> 7), j = (int)(i & 127);
  float c = fmaxf(cnt[nd], 1.f);
  long o = (((long)nd * TT + t) << 7) + j;
  float mean = sum[o] / c;
  float var  = sumsq[o] / c - mean * mean;
  float sd   = sqrtf(fmaxf(var, 0.f) + 1e-5f);
  float m = mx[o];
  if (m < -1e37f) m = 0.f;          // isolated node: segment_max was -inf
  float* row = pin + (long)nd * (4 * FIN);
  row[j]           = x[(long)nd * DD + j];
  row[FIN + j]     = mean;
  row[2 * FIN + j] = m;
  row[3 * FIN + j] = sd;
}

__global__ void gru_kernel(const float* gi, const float* gh, float* h) {
  long i = (long)blockIdx.x * blockDim.x + threadIdx.x;
  if (i >= (long)NN * DD) return;
  int nd = (int)(i >> 7), j = (int)(i & 127);
  const float* gir = gi + (long)nd * (3 * DD);
  const float* ghr = gh + (long)nd * (3 * DD);
  float r  = 1.f / (1.f + expf(-(gir[j] + ghr[j])));
  float z  = 1.f / (1.f + expf(-(gir[DD + j] + ghr[DD + j])));
  float nn = tanhf(gir[2 * DD + j] + r * ghr[2 * DD + j]);
  float hv = h[i];
  h[i] = (1.f - z) * nn + z * hv;
}

__global__ void mul_kernel(const float* a, const float* b, float* o, long n) {
  long i = (long)blockIdx.x * blockDim.x + threadIdx.x;
  if (i < n) o[i] = a[i] * b[i];
}

// ---------------------------------------------------------------------------
// Host-side launch helpers
// ---------------------------------------------------------------------------
static inline void launch_gemm3(hipStream_t st,
    const float* A0, const int* g0, int ld0, int K0,
    const float* A1, const int* g1, int ld1, int K1,
    const float* A2, const int* g2, int ld2, int K2,
    const float* W, int Nc, const float* bias,
    float* C, int ldc, int M, int relu) {
  GemmParams p;
  p.A0 = A0; p.g0 = g0; p.ld0 = ld0; p.K0 = K0;
  p.A1 = A1; p.g1 = g1; p.ld1 = ld1; p.K1 = K1;
  p.A2 = A2; p.g2 = g2; p.ld2 = ld2; p.K2 = K2;
  p.W = W; p.ldw = Nc; p.bias = bias; p.C = C; p.ldc = ldc;
  p.M = M; p.relu = relu;
  if ((Nc & 63) == 0) {
    dim3 grid(Nc / 64, (M + 127) / 128);
    wmma_gemm<64><<<grid, dim3(256), 0, st>>>(p);
  } else {
    dim3 grid(Nc / 32, (M + 127) / 128);
    wmma_gemm<32><<<grid, dim3(256), 0, st>>>(p);
  }
}

static inline void launch_gemm(hipStream_t st,
    const float* A, const int* g, int ld, int K,
    const float* W, int Nc, const float* bias,
    float* C, int ldc, int M, int relu) {
  launch_gemm3(st, A, g, ld, K, nullptr, nullptr, 0, 0, nullptr, nullptr, 0, 0,
               W, Nc, bias, C, ldc, M, relu);
}

static inline void launch_fill(hipStream_t st, float* p, float v, long n) {
  fill_kernel<<<dim3((unsigned)((n + 255) / 256)), dim3(256), 0, st>>>(p, v, n);
}

// ---------------------------------------------------------------------------
extern "C" void kernel_launch(void* const* d_in, const int* in_sizes, int n_in,
                              void* d_out, int out_size, void* d_ws, size_t ws_size,
                              hipStream_t stream) {
  (void)in_sizes; (void)n_in; (void)out_size; (void)ws_size;

  const float* x       = (const float*)d_in[0];
  const int*   ei      = (const int*)  d_in[1];
  const float* ea_in   = (const float*)d_in[2];
  const float* enc_W   = (const float*)d_in[3];
  const float* enc_b   = (const float*)d_in[4];
  const float* pre_W1  = (const float*)d_in[5];
  const float* pre_b1  = (const float*)d_in[6];
  const float* pre_W2  = (const float*)d_in[7];
  const float* pre_b2  = (const float*)d_in[8];
  const float* post_W1 = (const float*)d_in[9];
  const float* post_b1 = (const float*)d_in[10];
  const float* post_W2 = (const float*)d_in[11];
  const float* post_b2 = (const float*)d_in[12];
  const float* lin_W   = (const float*)d_in[13];
  const float* lin_b   = (const float*)d_in[14];
  const float* gru_Wi  = (const float*)d_in[15];
  const float* gru_Wh  = (const float*)d_in[16];
  const float* gru_bi  = (const float*)d_in[17];
  const float* gru_bh  = (const float*)d_in[18];
  const float* fcu_W1  = (const float*)d_in[19];
  const float* fcu_b1  = (const float*)d_in[20];
  const float* fcu_W2  = (const float*)d_in[21];
  const float* fcu_b2  = (const float*)d_in[22];
  const float* fcv_W1  = (const float*)d_in[23];
  const float* fcv_b1  = (const float*)d_in[24];
  const float* fcv_W2  = (const float*)d_in[25];
  const float* fcv_b2  = (const float*)d_in[26];
  const float* fc_W1   = (const float*)d_in[27];
  const float* fc_b1   = (const float*)d_in[28];
  const float* fc_W2   = (const float*)d_in[29];
  const float* fc_b2   = (const float*)d_in[30];
  const float* eu_W1   = (const float*)d_in[31];
  const float* eu_b1   = (const float*)d_in[32];
  const float* eu_W2   = (const float*)d_in[33];
  const float* eu_b2   = (const float*)d_in[34];

  const int* src = ei;        // edge_index[0]
  const int* dst = ei + EE;   // edge_index[1]

  // ---- workspace arena (bump allocator, 256B aligned) --------------------
  float* ws = (float*)d_ws;
  size_t off = 0;
  auto alloc = [&](size_t nf) {
    float* p = ws + off;
    off += (nf + 63) & ~(size_t)63;
    return p;
  };
  float* hbuf  = alloc((size_t)NN * DD);        // out == h (equal after GRU)
  float* EA    = alloc((size_t)EE * EDD);       // evolving edge_attr
  float* cnt   = alloc(NN);
  float* sum   = alloc((size_t)NN * TT * FIN);
  float* sumsq = alloc((size_t)NN * TT * FIN);
  float* mx    = alloc((size_t)NN * TT * FIN);
  float* ybuf  = alloc((size_t)NN * DD);
  float* mbuf  = alloc((size_t)NN * DD);
  float* MED   = alloc((size_t)CH * EDD);
  float* CB1   = alloc((size_t)CH * 256);
  float* CB2   = alloc((size_t)CH * 256);
  float* CB3   = alloc((size_t)CH * 256);
  // phase-aliased views (message/post/GRU phases never overlap edge phase)
  float* EBe = CB1;  float* HB  = CB2;  float* MB  = CB3;  // message phase
  float* PIN = CB1;  float* PB  = CB2;                     // post phase
  float* gib = CB1;  float* ghb = CB3;                     // GRU phase

  // ---- init --------------------------------------------------------------
  hipMemcpyAsync(hbuf, x, (size_t)NN * DD * sizeof(float),
                 hipMemcpyDeviceToDevice, stream);
  hipMemcpyAsync(EA, ea_in, (size_t)EE * EDD * sizeof(float),
                 hipMemcpyDeviceToDevice, stream);
  launch_fill(stream, cnt, 0.f, NN);
  count_kernel<<<dim3((EE + 255) / 256), dim3(256), 0, stream>>>(dst, cnt, EE);

  const long aggN = (long)NN * TT * FIN;

  for (int l = 0; l < LL; ++l) {
    // ---- PNA message + aggregate (chunked over edges) --------------------
    launch_fill(stream, sum,   0.f,    aggN);
    launch_fill(stream, sumsq, 0.f,    aggN);
    launch_fill(stream, mx,   -3e38f,  aggN);

    for (int s = 0; s < EE; s += CH) {
      int mc = (EE - s < CH) ? (EE - s) : CH;
      // e = edge_attr @ enc_W + enc_b        (mc x 64 -> mc x 128)
      launch_gemm(stream, EA + (size_t)s * EDD, nullptr, EDD, EDD,
                  enc_W + (size_t)l * EDD * FIN, FIN, enc_b + (size_t)l * FIN,
                  EBe, FIN, mc, 0);
      for (int t = 0; t < TT; ++t) {
        int lt = l * TT + t;
        // h = relu(x[dst]@Wi + x[src]@Wj + e@We + b1): one 3-source GEMM
        // over the contiguous 384x128 pre_W1[l][t]
        launch_gemm3(stream,
                     hbuf, dst + s, DD, FIN,
                     hbuf, src + s, DD, FIN,
                     EBe, nullptr, FIN, FIN,
                     pre_W1 + (size_t)lt * 3 * FIN * FIN, FIN,
                     pre_b1 + (size_t)lt * FIN,
                     HB, FIN, mc, 1);
        // msg = h @ W2 + b2
        launch_gemm(stream, HB, nullptr, FIN, FIN,
                    pre_W2 + (size_t)lt * FIN * FIN, FIN,
                    pre_b2 + (size_t)lt * FIN,
                    MB, FIN, mc, 0);
        // fused segment sum / sumsq / max
        long tot = (long)mc * FIN;
        aggregate_kernel<<<dim3((unsigned)((tot + 255) / 256)), dim3(256), 0,
                           stream>>>(MB, dst + s, mc, t, sum, sumsq, mx);
      }
    }

    // ---- post-MLP per task, then lin ------------------------------------
    for (int t = 0; t < TT; ++t) {
      int lt = l * TT + t;
      build_postin_kernel<<<dim3((NN * FIN + 255) / 256), dim3(256), 0,
                            stream>>>(hbuf, sum, sumsq, mx, cnt, PIN, t);
      launch_gemm(stream, PIN, nullptr, 4 * FIN, 4 * FIN,
                  post_W1 + (size_t)lt * 4 * FIN * FOUT, FOUT,
                  post_b1 + (size_t)lt * FOUT, PB, FOUT, NN, 1);
      launch_gemm(stream, PB, nullptr, FOUT, FOUT,
                  post_W2 + (size_t)lt * FOUT * FOUT, FOUT,
                  post_b2 + (size_t)lt * FOUT,
                  ybuf + t * FOUT, DD, NN, 0);   // writes y[:, t*32:(t+1)*32]
    }
    launch_gemm(stream, ybuf, nullptr, DD, DD,
                lin_W + (size_t)l * DD * DD, DD, lin_b + (size_t)l * DD,
                mbuf, DD, NN, 0);

    // ---- GRU -------------------------------------------------------------
    launch_gemm(stream, mbuf, nullptr, DD, DD, gru_Wi, 3 * DD, gru_bi,
                gib, 3 * DD, NN, 0);
    launch_gemm(stream, hbuf, nullptr, DD, DD, gru_Wh, 3 * DD, gru_bh,
                ghb, 3 * DD, NN, 0);
    gru_kernel<<<dim3((NN * DD + 255) / 256), dim3(256), 0, stream>>>(gib, ghb,
                                                                      hbuf);

    // ---- per-edge MLP chain + edge_attr update (L2-resident chunks) ------
    for (int s = 0; s < EE; s += CH) {
      int mc = (EE - s < CH) ? (EE - s) : CH;
      // fu = relu(relu(out[src]@W1+b1)@W2+b2)
      launch_gemm(stream, hbuf, src + s, DD, DD, fcu_W1, 2 * DD, fcu_b1,
                  CB1, 2 * DD, mc, 1);
      launch_gemm(stream, CB1, nullptr, 2 * DD, 2 * DD, fcu_W2, 2 * DD, fcu_b2,
                  CB2, 2 * DD, mc, 1);
      // fv = relu(relu(out[dst]@W1+b1)@W2+b2)
      launch_gemm(stream, hbuf, dst + s, DD, DD, fcv_W1, 2 * DD, fcv_b1,
                  CB1, 2 * DD, mc, 1);
      launch_gemm(stream, CB1, nullptr, 2 * DD, 2 * DD, fcv_W2, 2 * DD, fcv_b2,
                  CB3, 2 * DD, mc, 1);
      // prod = fu * fv
      mul_kernel<<<dim3((unsigned)(((long)mc * 2 * DD + 255) / 256)), dim3(256),
                   0, stream>>>(CB2, CB3, CB1, (long)mc * 2 * DD);
      // m_ed = relu(relu(prod@fc_W1+b1)@fc_W2+b2)
      launch_gemm(stream, CB1, nullptr, 2 * DD, 2 * DD, fc_W1, 2 * DD, fc_b1,
                  CB2, 2 * DD, mc, 1);
      launch_gemm(stream, CB2, nullptr, 2 * DD, 2 * DD, fc_W2, EDD, fc_b2,
                  MED, EDD, mc, 1);
      // edge_attr = relu([m_ed | edge_attr] @ eu_W1 + b1) @ eu_W2 + b2
      launch_gemm3(stream,
                   MED, nullptr, EDD, EDD,
                   EA + (size_t)s * EDD, nullptr, EDD, EDD,
                   nullptr, nullptr, 0, 0,
                   eu_W1, 2 * EDD, eu_b1, CB3, 2 * EDD, mc, 1);
      launch_gemm(stream, CB3, nullptr, 2 * EDD, 2 * EDD, eu_W2, EDD, eu_b2,
                  EA + (size_t)s * EDD, EDD, mc, 0);
    }
  }

  // ---- outputs: (out, edge_attr, out) ------------------------------------
  float* out = (float*)d_out;
  hipMemcpyAsync(out, hbuf, (size_t)NN * DD * sizeof(float),
                 hipMemcpyDeviceToDevice, stream);
  hipMemcpyAsync(out + (size_t)NN * DD, EA, (size_t)EE * EDD * sizeof(float),
                 hipMemcpyDeviceToDevice, stream);
  hipMemcpyAsync(out + (size_t)NN * DD + (size_t)EE * EDD, hbuf,
                 (size_t)NN * DD * sizeof(float), hipMemcpyDeviceToDevice,
                 stream);
}
